// RingEncoder_79585743994945
// MI455X (gfx1250) — matrix operand order repaired and checked by reading, not compiled, not run on hardware
//
#include <hip/hip_runtime.h>
#include <hip/hip_bf16.h>

// ---------------------------------------------------------------------------
// RingEncoder forward for MI455X (gfx1250, wave32, WMMA).
// All dense GEMMs use v_wmma_f32_16x16x32_f16 (f16 A/B, f32 accumulate).
// ---------------------------------------------------------------------------

typedef _Float16 h16v __attribute__((ext_vector_type(16)));
typedef _Float16 h8v  __attribute__((ext_vector_type(8)));
typedef float    f8v  __attribute__((ext_vector_type(8)));

#define NB 8
#define NCIN 5
#define NN 8192
#define NR 16
#define NP (NB * NN)
#define BN_EPS 1e-5f

// Order-preserving float <-> uint mapping for deterministic atomic max.
__device__ __forceinline__ unsigned fenc(float f) {
  unsigned u = __float_as_uint(f);
  return (u & 0x80000000u) ? ~u : (u | 0x80000000u);
}
__device__ __forceinline__ float fdec(unsigned u) {
  unsigned b = (u & 0x80000000u) ? (u ^ 0x80000000u) : ~u;
  return __uint_as_float(b);
}

// ---------------------------------------------------------------------------
// Generic WMMA GEMM:  Z[p, o] = sum_c X[p,c] * W[o,c] (+ bias[o])
// X: fp16 [P, Cin] row-major.  W: fp16 [Cout, Cin] row-major (B = W^T).
// One 16x16 output tile per wave; 8 waves per block.
// wBatchStride != 0 -> per-batch weights (W + (p/NN)*wBatchStride), used for
// the per-batch h @ trans_feat GEMM (weights pre-transposed to [Cout,Cin]).
// ---------------------------------------------------------------------------
__global__ void k_wmma_gemm(const _Float16* __restrict__ X,
                            const _Float16* __restrict__ W,
                            const float* __restrict__ bias,
                            _Float16* __restrict__ Zh,
                            float* __restrict__ Zf,
                            int Cin, int Cout, int wBatchStride) {
  const int wave = threadIdx.x >> 5;
  const int lane = threadIdx.x & 31;
  const long tile = (long)blockIdx.x * 8 + wave;
  const int colTiles = Cout >> 4;
  const int rowTile = (int)(tile / colTiles);
  const int colTile = (int)(tile % colTiles);
  const int p0 = rowTile << 4;
  const int n0 = colTile << 4;
  const _Float16* Wb = W + (wBatchStride ? (long)(p0 >> 13) * wBatchStride : 0);

  f8v acc = {0.f, 0.f, 0.f, 0.f, 0.f, 0.f, 0.f, 0.f};

  // A fragment: lane -> M = lane&15; halves 0..7 at K = kk + (lane<16?0:8),
  // halves 8..15 at K = kk + 16 + (lane<16?0:8).
  const int mrow = p0 + (lane & 15);
  const int koffA = (lane < 16) ? 0 : 8;
  // B fragment: lane -> N = lane&15; halves 0..7 at K = kk + (lane<16?0:8),
  // halves 8..15 at K = kk + 16 + (lane<16?0:8).
  const int nB = n0 + (lane & 15);
  const int koffB = (lane < 16) ? 0 : 8;

  for (int kk = 0; kk < Cin; kk += 32) {
    union { h16v v; h8v h[2]; } a, b;
    a.h[0] = *(const h8v*)(X + (long)mrow * Cin + kk + koffA);
    a.h[1] = *(const h8v*)(X + (long)mrow * Cin + kk + 16 + koffA);
    b.h[0] = *(const h8v*)(Wb + (long)nB * Cin + kk + koffB);
    b.h[1] = *(const h8v*)(Wb + (long)nB * Cin + kk + 16 + koffB);
    acc = __builtin_amdgcn_wmma_f32_16x16x32_f16(
        false, a.v, false, b.v, (short)0, acc, false, false);
  }

  // C/D layout: VGPR r -> M = r + (lane<16 ? 0 : 8), N = lane&15.
  const int col = n0 + (lane & 15);
  const float bv = bias ? bias[col] : 0.0f;
  const int rbase = p0 + ((lane < 16) ? 0 : 8);
#pragma unroll
  for (int r = 0; r < 8; ++r) {
    float v = acc[r] + bv;
    long idx = (long)(rbase + r) * Cout + col;
    if (Zh) Zh[idx] = (_Float16)v;
    if (Zf) Zf[idx] = v;
  }
}

// ---------------------------------------------------------------------------
// Deterministic two-stage per-channel sum / sumsq over P rows.
// ---------------------------------------------------------------------------
__global__ void k_stats_partial(const _Float16* __restrict__ Z,
                                float* __restrict__ part,
                                int C, int rowsPerChunk) {
  const int c = blockIdx.x * 64 + (threadIdx.x & 63);
  const int sub = threadIdx.x >> 6;  // 0..3
  const int r0 = blockIdx.y * rowsPerChunk;
  float s = 0.f, q = 0.f;
  for (int r = sub; r < rowsPerChunk; r += 4) {
    float v = (float)Z[(long)(r0 + r) * C + c];
    s += v;
    q += v * v;
  }
  __shared__ float ls[256], lq[256];
  ls[threadIdx.x] = s;
  lq[threadIdx.x] = q;
  __syncthreads();
  if (sub == 0) {
    int t = threadIdx.x;
    s = ls[t] + ls[t + 64] + ls[t + 128] + ls[t + 192];
    q = lq[t] + lq[t + 64] + lq[t + 128] + lq[t + 192];
    part[((long)blockIdx.y * C + c) * 2 + 0] = s;
    part[((long)blockIdx.y * C + c) * 2 + 1] = q;
  }
}

__global__ void k_stats_finalize(const float* __restrict__ part,
                                 const float* __restrict__ g,
                                 const float* __restrict__ be,
                                 float* __restrict__ ss,
                                 int C, int nChunk, float invCnt) {
  const int c = blockIdx.x * 256 + threadIdx.x;
  if (c >= C) return;
  float s = 0.f, q = 0.f;
  for (int k = 0; k < nChunk; ++k) {
    s += part[((long)k * C + c) * 2 + 0];
    q += part[((long)k * C + c) * 2 + 1];
  }
  float m = s * invCnt;
  float var = fmaxf(q * invCnt - m * m, 0.f);
  float sc = g[c] * rsqrtf(var + BN_EPS);
  ss[c * 2 + 0] = sc;
  ss[c * 2 + 1] = be[c] - m * sc;
}

// Elementwise BN-apply (+ReLU) -> fp16 activations.
__global__ void k_apply(const _Float16* __restrict__ Z,
                        const float* __restrict__ ss,
                        _Float16* __restrict__ A,
                        long total, int C, int relu) {
  long i = (long)blockIdx.x * blockDim.x + threadIdx.x;
  if (i >= total) return;
  int c = (int)(i % C);
  float v = (float)Z[i] * ss[c * 2] + ss[c * 2 + 1];
  if (relu) v = fmaxf(v, 0.f);
  A[i] = (_Float16)v;
}

// BN-apply (+opt ReLU) on [P,1024] fused with per-batch max over N.
// grid: 32 blocks per batch (256 rows each), block 256 (4 channels / thread).
__global__ void k_apply_max(const _Float16* __restrict__ Z,
                            const float* __restrict__ ss,
                            unsigned* __restrict__ umax, int relu) {
  const int b = blockIdx.x >> 5;
  const int r0 = b * NN + (blockIdx.x & 31) * 256;
  const int c0 = threadIdx.x * 4;
  float s0 = ss[(c0 + 0) * 2], h0 = ss[(c0 + 0) * 2 + 1];
  float s1 = ss[(c0 + 1) * 2], h1 = ss[(c0 + 1) * 2 + 1];
  float s2 = ss[(c0 + 2) * 2], h2 = ss[(c0 + 2) * 2 + 1];
  float s3 = ss[(c0 + 3) * 2], h3 = ss[(c0 + 3) * 2 + 1];
  float m0 = -3.0e38f, m1 = -3.0e38f, m2 = -3.0e38f, m3 = -3.0e38f;
  for (int r = 0; r < 256; ++r) {
    const _Float16* zp = Z + (long)(r0 + r) * 1024 + c0;
    float v0 = (float)zp[0] * s0 + h0;
    float v1 = (float)zp[1] * s1 + h1;
    float v2 = (float)zp[2] * s2 + h2;
    float v3 = (float)zp[3] * s3 + h3;
    if (relu) {
      v0 = fmaxf(v0, 0.f); v1 = fmaxf(v1, 0.f);
      v2 = fmaxf(v2, 0.f); v3 = fmaxf(v3, 0.f);
    }
    m0 = fmaxf(m0, v0); m1 = fmaxf(m1, v1);
    m2 = fmaxf(m2, v2); m3 = fmaxf(m3, v3);
  }
  atomicMax(&umax[b * 1024 + c0 + 0], fenc(m0));
  atomicMax(&umax[b * 1024 + c0 + 1], fenc(m1));
  atomicMax(&umax[b * 1024 + c0 + 2], fenc(m2));
  atomicMax(&umax[b * 1024 + c0 + 3], fenc(m3));
}

__global__ void k_decode(const unsigned* __restrict__ u, float* __restrict__ f, int n) {
  int i = blockIdx.x * 256 + threadIdx.x;
  if (i < n) f[i] = fdec(u[i]);
}

__global__ void k_f2h(const float* __restrict__ s, _Float16* __restrict__ d, int n) {
  int i = blockIdx.x * 256 + threadIdx.x;
  if (i < n) d[i] = (_Float16)s[i];
}

// ---------------------------------------------------------------------------
// 5 -> 64 conv (VALU; K too small for WMMA). Reads x [B,5,N] directly.
// grid NP/64, block 256 (64 points x 4 threads x 16 outputs).
// ---------------------------------------------------------------------------
__global__ void k_lin5_stn(const float* __restrict__ x,
                           const float* __restrict__ W,
                           const float* __restrict__ bias,
                           _Float16* __restrict__ Z) {
  __shared__ float sW[384];
  for (int i = threadIdx.x; i < 320; i += 256) sW[i] = W[i];
  for (int i = threadIdx.x; i < 64; i += 256) sW[320 + i] = bias[i];
  __syncthreads();
  const int p = blockIdx.x * 64 + (threadIdx.x >> 2);
  const int b = p >> 13, n = p & 8191;
  float v[5];
#pragma unroll
  for (int c = 0; c < 5; ++c) v[c] = x[((long)b * 5 + c) * NN + n];
  const int o0 = (threadIdx.x & 3) * 16;
#pragma unroll
  for (int o = 0; o < 16; ++o) {
    int oo = o0 + o;
    float acc = sW[320 + oo];
#pragma unroll
    for (int c = 0; c < 5; ++c) acc += v[c] * sW[oo * 5 + c];
    Z[(long)p * 64 + oo] = (_Float16)acc;
  }
}

// Same, fused with the 3x3 input transform (xyz' = xyz @ trans[b]).
__global__ void k_lin5_c1(const float* __restrict__ x,
                          const float* __restrict__ trans,
                          const float* __restrict__ W,
                          const float* __restrict__ bias,
                          _Float16* __restrict__ Z) {
  __shared__ float sW[384];
  for (int i = threadIdx.x; i < 320; i += 256) sW[i] = W[i];
  for (int i = threadIdx.x; i < 64; i += 256) sW[320 + i] = bias[i];
  __syncthreads();
  const int p = blockIdx.x * 64 + (threadIdx.x >> 2);
  const int b = p >> 13, n = p & 8191;
  float xv[5];
#pragma unroll
  for (int c = 0; c < 5; ++c) xv[c] = x[((long)b * 5 + c) * NN + n];
  const float* t = trans + b * 9;
  float v[5];
  v[0] = xv[0] * t[0] + xv[1] * t[3] + xv[2] * t[6];
  v[1] = xv[0] * t[1] + xv[1] * t[4] + xv[2] * t[7];
  v[2] = xv[0] * t[2] + xv[1] * t[5] + xv[2] * t[8];
  v[3] = xv[3];
  v[4] = xv[4];
  const int o0 = (threadIdx.x & 3) * 16;
#pragma unroll
  for (int o = 0; o < 16; ++o) {
    int oo = o0 + o;
    float acc = sW[320 + oo];
#pragma unroll
    for (int c = 0; c < 5; ++c) acc += v[c] * sW[oo * 5 + c];
    Z[(long)p * 64 + oo] = (_Float16)acc;
  }
}

// ---------------------------------------------------------------------------
// Tiny FC layers (batch = 8): one block per output channel, BN over batch.
// ---------------------------------------------------------------------------
__global__ void k_fc_bn_relu(const float* __restrict__ X,
                             const float* __restrict__ W,
                             const float* __restrict__ bias,
                             const float* __restrict__ g,
                             const float* __restrict__ be,
                             float* __restrict__ Y, int Cin, int Cout) {
  const int o = blockIdx.x, t = threadIdx.x;
  float acc[8] = {0.f, 0.f, 0.f, 0.f, 0.f, 0.f, 0.f, 0.f};
  for (int k = t; k < Cin; k += 64) {
    float w = W[(long)o * Cin + k];
#pragma unroll
    for (int b = 0; b < 8; ++b) acc[b] += X[b * Cin + k] * w;
  }
  __shared__ float ls[512];
#pragma unroll
  for (int b = 0; b < 8; ++b) ls[b * 64 + t] = acc[b];
  __syncthreads();
  for (int s = 32; s > 0; s >>= 1) {
    if (t < s)
      for (int b = 0; b < 8; ++b) ls[b * 64 + t] += ls[b * 64 + t + s];
    __syncthreads();
  }
  if (t == 0) {
    float z[8], m = 0.f;
#pragma unroll
    for (int b = 0; b < 8; ++b) { z[b] = ls[b * 64] + bias[o]; m += z[b]; }
    m *= 0.125f;
    float var = 0.f;
#pragma unroll
    for (int b = 0; b < 8; ++b) var += (z[b] - m) * (z[b] - m);
    var *= 0.125f;
    float sc = g[o] * rsqrtf(var + BN_EPS);
    float sh = be[o] - m * sc;
#pragma unroll
    for (int b = 0; b < 8; ++b) Y[b * Cout + o] = fmaxf(z[b] * sc + sh, 0.f);
  }
}

__global__ void k_fc_eye(const float* __restrict__ X,
                         const float* __restrict__ W,
                         const float* __restrict__ bias,
                         float* __restrict__ Y, int Cin, int KK, int k) {
  const int o = blockIdx.x, t = threadIdx.x;
  float acc[8] = {0.f, 0.f, 0.f, 0.f, 0.f, 0.f, 0.f, 0.f};
  for (int kk = t; kk < Cin; kk += 64) {
    float w = W[(long)o * Cin + kk];
#pragma unroll
    for (int b = 0; b < 8; ++b) acc[b] += X[b * Cin + kk] * w;
  }
  __shared__ float ls[512];
#pragma unroll
  for (int b = 0; b < 8; ++b) ls[b * 64 + t] = acc[b];
  __syncthreads();
  for (int s = 32; s > 0; s >>= 1) {
    if (t < s)
      for (int b = 0; b < 8; ++b) ls[b * 64 + t] += ls[b * 64 + t + s];
    __syncthreads();
  }
  if (t == 0) {
    float e = (o % (k + 1) == 0) ? 1.f : 0.f;
#pragma unroll
    for (int b = 0; b < 8; ++b) Y[b * KK + o] = ls[b * 64] + bias[o] + e;
  }
}

// trans_feat [8,64,64] fp32 -> fp16, transposed to [e][d] so the WMMA GEMM can
// consume it as a [Cout,Cin] weight matrix.
__global__ void k_tf16t(const float* __restrict__ tf, _Float16* __restrict__ T) {
  int i = blockIdx.x * 256 + threadIdx.x;
  if (i >= NB * 4096) return;
  int b = i >> 12, r = i & 4095, e = r >> 6, d = r & 63;
  T[i] = (_Float16)tf[(b << 12) + d * 64 + e];
}

// ---------------------------------------------------------------------------
// Ring experts: y = h @ W[ring]^T + b[ring] + masked BN stats (per-block
// partials, deterministic final reduce). grid NP/64, block 256.
// ---------------------------------------------------------------------------
__global__ void k_ring_gemm(const float* __restrict__ Hf,
                            const int* __restrict__ ring,
                            const float* __restrict__ W,
                            const float* __restrict__ bias,
                            float* __restrict__ Y,
                            float* __restrict__ part) {
  __shared__ float hT[64][65];
  __shared__ float sSum[2048], sSq[2048];
  __shared__ int sCnt[16];
  const int p0 = blockIdx.x * 64;
  for (int i = threadIdx.x; i < 4096; i += 256)
    hT[i >> 6][i & 63] = Hf[(long)p0 * 64 + i];
  for (int i = threadIdx.x; i < 2048; i += 256) { sSum[i] = 0.f; sSq[i] = 0.f; }
  if (threadIdx.x < 16) sCnt[threadIdx.x] = 0;
  __syncthreads();
  const int pl = threadIdx.x & 63;
  const int og = threadIdx.x >> 6;  // 0..3 -> 32 outputs each
  const int p = p0 + pl;
  const int r = ring[p];
  if (og == 0) atomicAdd(&sCnt[r], 1);
  for (int o = og * 32; o < og * 32 + 32; ++o) {
    float acc = bias[r * 128 + o];
    const float* w = W + (long)(r * 128 + o) * 64;
#pragma unroll 8
    for (int c = 0; c < 64; ++c) acc += hT[pl][c] * w[c];
    Y[(long)p * 128 + o] = acc;
    atomicAdd(&sSum[r * 128 + o], acc);
    atomicAdd(&sSq[r * 128 + o], acc * acc);
  }
  __syncthreads();
  float* pb = part + (long)blockIdx.x * 4112;
  for (int i = threadIdx.x; i < 2048; i += 256) {
    pb[i] = sSum[i];
    pb[2048 + i] = sSq[i];
  }
  if (threadIdx.x < 16) pb[4096 + threadIdx.x] = (float)sCnt[threadIdx.x];
}

__global__ void k_ring_stats(const float* __restrict__ part,
                             const float* __restrict__ g,
                             const float* __restrict__ be,
                             float* __restrict__ ss, int nBlk) {
  const int i = blockIdx.x * 256 + threadIdx.x;  // r*128 + c, 2048 total
  if (i >= 2048) return;
  const int r = i >> 7;
  float s = 0.f, q = 0.f, cnt = 0.f;
  for (int k = 0; k < nBlk; ++k) {
    const float* pb = part + (long)k * 4112;
    s += pb[i];
    q += pb[2048 + i];
    cnt += pb[4096 + r];
  }
  float ic = 1.f / fmaxf(cnt, 1.f);
  float m = s * ic;
  float var = fmaxf(q * ic - m * m, 0.f);
  float sc = g[i] * rsqrtf(var + BN_EPS);
  ss[i * 2 + 0] = sc;
  ss[i * 2 + 1] = be[i] - m * sc;
}

// Per-(batch, ring, channel) max of normalized ring outputs.
// grid NP/256 (256-point blocks never cross a batch), block 256.
__global__ void k_ring_max(const float* __restrict__ Y,
                           const int* __restrict__ ring,
                           const float* __restrict__ ss,
                           unsigned* __restrict__ ymax) {
  __shared__ unsigned um[2048];
  for (int i = threadIdx.x; i < 2048; i += 256) um[i] = 0u;
  __syncthreads();
  const int p0 = blockIdx.x * 256;
  const int c = threadIdx.x & 127;
  const int rsub = threadIdx.x >> 7;  // 0..1
  for (int r = rsub; r < 256; r += 2) {
    int p = p0 + r;
    int rg = ring[p];
    int ci = rg * 128 + c;
    float v = Y[(long)p * 128 + c] * ss[ci * 2] + ss[ci * 2 + 1];
    atomicMax(&um[ci], fenc(v));
  }
  __syncthreads();
  const int b = p0 >> 13;
  for (int i = threadIdx.x; i < 2048; i += 256) {
    unsigned u = um[i];
    if (u) atomicMax(&ymax[b * 2048 + i], u);
  }
}

// Assemble out[b, ch, n]: [x1 (1024) | x2 (128) | h (64)].
__global__ void k_final(const float* __restrict__ x1f,
                        const unsigned* __restrict__ ymax,
                        const float* __restrict__ Hf,
                        const int* __restrict__ ring,
                        float* __restrict__ out) {
  const int n = blockIdx.x * 256 + threadIdx.x;
  const int ch = blockIdx.y;
  const int b = blockIdx.z;
  const int p = b * NN + n;
  float v;
  if (ch < 1024) {
    v = x1f[b * 1024 + ch];
  } else if (ch < 1152) {
    int rg = ring[p];
    v = fdec(ymax[(b * 16 + rg) * 128 + (ch - 1024)]);
  } else {
    v = Hf[(long)p * 64 + (ch - 1152)];
  }
  out[((long)b * 1216 + ch) * NN + n] = v;
}

// ---------------------------------------------------------------------------
// Host side
// ---------------------------------------------------------------------------
struct StnPtrs {
  const float *c1W, *c1b, *c2W, *c2b, *c3W, *c3b;
  const float *f1W, *f1b, *f2W, *f2b, *f3W, *f3b;
  const float *g1, *b1, *g2, *b2, *g3, *b3, *g4, *b4, *g5, *b5;
};

static StnPtrs load_stn(void* const* d, int base) {
  StnPtrs s;
  s.c1W = (const float*)d[base + 0];  s.c1b = (const float*)d[base + 1];
  s.c2W = (const float*)d[base + 2];  s.c2b = (const float*)d[base + 3];
  s.c3W = (const float*)d[base + 4];  s.c3b = (const float*)d[base + 5];
  s.f1W = (const float*)d[base + 6];  s.f1b = (const float*)d[base + 7];
  s.f2W = (const float*)d[base + 8];  s.f2b = (const float*)d[base + 9];
  s.f3W = (const float*)d[base + 10]; s.f3b = (const float*)d[base + 11];
  s.g1 = (const float*)d[base + 12];  s.b1 = (const float*)d[base + 13];
  s.g2 = (const float*)d[base + 14];  s.b2 = (const float*)d[base + 15];
  s.g3 = (const float*)d[base + 16];  s.b3 = (const float*)d[base + 17];
  s.g4 = (const float*)d[base + 18];  s.b4 = (const float*)d[base + 19];
  s.g5 = (const float*)d[base + 20];  s.b5 = (const float*)d[base + 21];
  return s;
}

extern "C" void kernel_launch(void* const* d_in, const int* in_sizes, int n_in,
                              void* d_out, int out_size, void* d_ws, size_t ws_size,
                              hipStream_t stream) {
  (void)in_sizes; (void)n_in; (void)out_size; (void)ws_size;

  const float* x = (const float*)d_in[0];
  const int* ring = (const int*)d_in[1];
  StnPtrs S = load_stn(d_in, 2);    // stn
  StnPtrs F = load_stn(d_in, 24);   // fstn
  const float* c1W = (const float*)d_in[46];
  const float* c1b = (const float*)d_in[47];
  const float* g1m = (const float*)d_in[48];
  const float* b1m = (const float*)d_in[49];
  const float* gp_c1W = (const float*)d_in[50];
  const float* gp_c1b = (const float*)d_in[51];
  const float* gp_g1 = (const float*)d_in[52];
  const float* gp_b1 = (const float*)d_in[53];
  const float* gp_c2W = (const float*)d_in[54];
  const float* gp_c2b = (const float*)d_in[55];
  const float* gp_g2 = (const float*)d_in[56];
  const float* gp_b2 = (const float*)d_in[57];
  const float* rpW = (const float*)d_in[58];
  const float* rpb = (const float*)d_in[59];
  const float* rpg = (const float*)d_in[60];
  const float* rpbe = (const float*)d_in[61];

  float* out = (float*)d_out;
  float* transOut = out + (size_t)NB * 1216 * NN;  // [B,3,3]
  float* tfOut = transOut + NB * 9;                // [B,64,64]

  // Workspace bump allocator (256B aligned).
  char* wp = (char*)d_ws;
  auto alloc = [&](size_t bytes) -> void* {
    void* p = (void*)wp;
    wp += (bytes + 255) & ~(size_t)255;
    return p;
  };
  _Float16* Zbig = (_Float16*)alloc((size_t)NP * 1024 * 2);
  _Float16* Z128 = (_Float16*)alloc((size_t)NP * 128 * 2);
  _Float16* A128 = (_Float16*)alloc((size_t)NP * 128 * 2);
  _Float16* Z64  = (_Float16*)alloc((size_t)NP * 64 * 2);
  _Float16* H1h  = (_Float16*)alloc((size_t)NP * 64 * 2);
  _Float16* A64b = (_Float16*)alloc((size_t)NP * 64 * 2);
  float*    Hf   = (float*)alloc((size_t)NP * 64 * 4);
  _Float16* Hf16 = (_Float16*)alloc((size_t)NP * 64 * 2);
  float*    Yring = (float*)alloc((size_t)NP * 128 * 4);
  float*    part = (float*)alloc((size_t)64 * 1024 * 2 * 4);
  float*    ss   = (float*)alloc(1024 * 2 * 4);
  unsigned* umax = (unsigned*)alloc(NB * 1024 * 4);
  float*    smaxf = (float*)alloc(NB * 1024 * 4);
  float*    fmaxf_ = (float*)alloc(NB * 1024 * 4);
  float*    x1f  = (float*)alloc(NB * 1024 * 4);
  float*    t512 = (float*)alloc(NB * 512 * 4);
  float*    t256 = (float*)alloc(NB * 256 * 4);
  _Float16* T16t = (_Float16*)alloc((size_t)NB * 4096 * 2);
  float*    rpart = (float*)alloc((size_t)1024 * 4112 * 4);
  float*    rss  = (float*)alloc(2048 * 2 * 4);
  unsigned* ymaxu = (unsigned*)alloc(NB * 2048 * 4);
  _Float16* w_s2 = (_Float16*)alloc(128 * 64 * 2);
  _Float16* w_s3 = (_Float16*)alloc(1024 * 128 * 2);
  _Float16* w_f1 = (_Float16*)alloc(64 * 64 * 2);
  _Float16* w_f2 = (_Float16*)alloc(128 * 64 * 2);
  _Float16* w_f3 = (_Float16*)alloc(1024 * 128 * 2);
  _Float16* w_g1 = (_Float16*)alloc(128 * 64 * 2);
  _Float16* w_g2 = (_Float16*)alloc(1024 * 128 * 2);

  auto f2h = [&](const float* s, _Float16* d, int n) {
    k_f2h<<<(n + 255) / 256, 256, 0, stream>>>(s, d, n);
  };
  auto gemm = [&](const _Float16* X, const _Float16* Wm, const float* bias,
                  _Float16* Zh, float* Zf, int Cin, int Cout, int wStride) {
    int blocks = (NP / 16) * (Cout / 16) / 8;
    k_wmma_gemm<<<blocks, 256, 0, stream>>>(X, Wm, bias, Zh, Zf, Cin, Cout, wStride);
  };
  auto stats = [&](const _Float16* Z, int C, const float* g, const float* be) {
    dim3 gs(C / 64, 64);
    k_stats_partial<<<gs, 256, 0, stream>>>(Z, part, C, NP / 64);
    k_stats_finalize<<<(C + 255) / 256, 256, 0, stream>>>(part, g, be, ss, C, 64,
                                                          1.0f / (float)NP);
  };
  auto apply = [&](const _Float16* Z, _Float16* A, int C, int relu) {
    long total = (long)NP * C;
    k_apply<<<(unsigned)(total / 256), 256, 0, stream>>>(Z, ss, A, total, C, relu);
  };
  auto fc_chain = [&](const float* inMax, const StnPtrs& sp, float* dst, int k) {
    k_fc_bn_relu<<<512, 64, 0, stream>>>(inMax, sp.f1W, sp.f1b, sp.g4, sp.b4,
                                         t512, 1024, 512);
    k_fc_bn_relu<<<256, 64, 0, stream>>>(t512, sp.f2W, sp.f2b, sp.g5, sp.b5,
                                         t256, 512, 256);
    k_fc_eye<<<k * k, 64, 0, stream>>>(t256, sp.f3W, sp.f3b, dst, 256, k * k, k);
  };

  // Weight fp32 -> fp16 conversions.
  f2h(S.c2W, w_s2, 128 * 64);   f2h(S.c3W, w_s3, 1024 * 128);
  f2h(F.c1W, w_f1, 64 * 64);    f2h(F.c2W, w_f2, 128 * 64);
  f2h(F.c3W, w_f3, 1024 * 128);
  f2h(gp_c1W, w_g1, 128 * 64);  f2h(gp_c2W, w_g2, 1024 * 128);

  // ---- Input STN (k=3) ----
  k_lin5_stn<<<NP / 64, 256, 0, stream>>>(x, S.c1W, S.c1b, Z64);
  stats(Z64, 64, S.g1, S.b1);
  apply(Z64, H1h, 64, 1);  // reuse H1h as stn activation A1
  gemm(H1h, w_s2, S.c2b, Z128, nullptr, 64, 128, 0);
  stats(Z128, 128, S.g2, S.b2);
  apply(Z128, A128, 128, 1);
  gemm(A128, w_s3, S.c3b, Zbig, nullptr, 128, 1024, 0);
  stats(Zbig, 1024, S.g3, S.b3);
  hipMemsetAsync(umax, 0, NB * 1024 * 4, stream);
  k_apply_max<<<256, 256, 0, stream>>>(Zbig, ss, umax, 1);
  k_decode<<<32, 256, 0, stream>>>(umax, smaxf, NB * 1024);
  fc_chain(smaxf, S, transOut, 3);

  // ---- c1 with fused xyz transform ----
  k_lin5_c1<<<NP / 64, 256, 0, stream>>>(x, transOut, c1W, c1b, Z64);
  stats(Z64, 64, g1m, b1m);
  apply(Z64, H1h, 64, 1);  // H1h = h before feature transform

  // ---- Feature STN (k=64) ----
  gemm(H1h, w_f1, F.c1b, Z64, nullptr, 64, 64, 0);
  stats(Z64, 64, F.g1, F.b1);
  apply(Z64, A64b, 64, 1);
  gemm(A64b, w_f2, F.c2b, Z128, nullptr, 64, 128, 0);
  stats(Z128, 128, F.g2, F.b2);
  apply(Z128, A128, 128, 1);
  gemm(A128, w_f3, F.c3b, Zbig, nullptr, 128, 1024, 0);
  stats(Zbig, 1024, F.g3, F.b3);
  hipMemsetAsync(umax, 0, NB * 1024 * 4, stream);
  k_apply_max<<<256, 256, 0, stream>>>(Zbig, ss, umax, 1);
  k_decode<<<32, 256, 0, stream>>>(umax, fmaxf_, NB * 1024);
  fc_chain(fmaxf_, F, tfOut, 64);
  k_tf16t<<<(NB * 4096 + 255) / 256, 256, 0, stream>>>(tfOut, T16t);

  // ---- h = h @ trans_feat (per-batch WMMA GEMM, transposed fp16 weights) ----
  gemm(H1h, T16t, nullptr, Hf16, Hf, 64, 64, 4096);

  // ---- Global pooler ----
  gemm(Hf16, w_g1, gp_c1b, Z128, nullptr, 64, 128, 0);
  stats(Z128, 128, gp_g1, gp_b1);
  apply(Z128, A128, 128, 1);
  gemm(A128, w_g2, gp_c2b, Zbig, nullptr, 128, 1024, 0);
  stats(Zbig, 1024, gp_g2, gp_b2);
  hipMemsetAsync(umax, 0, NB * 1024 * 4, stream);
  k_apply_max<<<256, 256, 0, stream>>>(Zbig, ss, umax, 0);  // no ReLU
  k_decode<<<32, 256, 0, stream>>>(umax, x1f, NB * 1024);

  // ---- Ring pooler ----
  k_ring_gemm<<<NP / 64, 256, 0, stream>>>(Hf, ring, rpW, rpb, Yring, rpart);
  k_ring_stats<<<8, 256, 0, stream>>>(rpart, rpg, rpbe, rss, NP / 64);
  hipMemsetAsync(ymaxu, 0, NB * 2048 * 4, stream);
  k_ring_max<<<NP / 256, 256, 0, stream>>>(Yring, ring, rss, ymaxu);

  // ---- Final concat [B,1216,N] ----
  k_final<<<dim3(NN / 256, 1216, NB), 256, 0, stream>>>(x1f, ymaxu, Hf, ring, out);
}
